// GNNLayer_78039555768490
// MI455X (gfx1250) — compile-verified, matrix operand order checked by reading, not run on hardware
//
#include <hip/hip_runtime.h>

// Problem constants (from reference)
#define N_NODES 50000
#define E_EDGES 250000
#define DIN     256
#define DOUT    256
#define OH      16
#define KPAD    288   // (DIN + OH)=272 padded up to 9*32 for WMMA K-steps

typedef __bf16 bf16;
typedef bf16  v16bf  __attribute__((ext_vector_type(16)));
typedef bf16  bf16x4 __attribute__((ext_vector_type(4)));
typedef float v8f    __attribute__((ext_vector_type(8)));

// 16-bit A/B fragment: 16 bf16 per lane = 32 bytes = two b128 loads.
// ISA layout (CDNA5 ISA): lane m=lane&15, kb=(lane>>4)*8;
// elements [0..7] = K kb..kb+7, elements [8..15] = K kb+16..kb+23.
union FragAB {
  v16bf v;
  uint4 u[2];
};

__device__ __forceinline__ FragAB load_frag(const bf16* __restrict__ p) {
  FragAB f;
  f.u[0] = *reinterpret_cast<const uint4*>(p);        // K .. K+7
  f.u[1] = *reinterpret_cast<const uint4*>(p + 16);   // K+16 .. K+23
  return f;
}

__device__ __forceinline__ v8f wmma_bf16(const FragAB& a, const FragAB& b, v8f c) {
  return __builtin_amdgcn_wmma_f32_16x16x32_bf16(false, a.v, false, b.v,
                                                 (short)0, c, false, false);
}

__device__ __forceinline__ float sigmoidf_fast(float x) {
  return 1.0f / (1.0f + __expf(-x));
}

// ---------------- pack kernels ----------------

// fp32 -> bf16, 4 elements per thread (coalesced b128 read, b64 write)
__global__ void pack_h_bf16(const float* __restrict__ src, bf16* __restrict__ dst, int n4) {
  int i = blockIdx.x * blockDim.x + threadIdx.x;
  if (i >= n4) return;
  float4 f = reinterpret_cast<const float4*>(src)[i];
  bf16x4 o;
  o.x = (bf16)f.x; o.y = (bf16)f.y; o.z = (bf16)f.z; o.w = (bf16)f.w;
  reinterpret_cast<bf16x4*>(dst)[i] = o;
}

// W is [K, Nc] fp32 row-major; produce W^T [Nc, KD] bf16 with zero pad for k>=K.
__global__ void pack_wT_bf16(const float* __restrict__ src, bf16* __restrict__ dst,
                             int K, int KD, int Nc) {
  int i = blockIdx.x * blockDim.x + threadIdx.x;
  int total = Nc * KD;
  if (i >= total) return;
  int n = i / KD;
  int k = i - n * KD;
  float v = (k < K) ? src[(size_t)k * Nc + n] : 0.0f;
  dst[i] = (bf16)v;
}

// ---------------- node linear: out = h @ W_lin + b ----------------
// One wave computes a 16x32 output tile (two 16x16 WMMA column sub-tiles),
// so each h-row fragment is loaded once per 32 output columns.

__global__ void __launch_bounds__(256)
node_linear_wmma(const bf16* __restrict__ hbf,   // [N, DIN] bf16
                 const bf16* __restrict__ wT,    // [DOUT, DIN] bf16 (W^T)
                 const float* __restrict__ bias, // [DOUT]
                 float* __restrict__ out) {      // [N, DOUT] fp32
  const int wave = (blockIdx.x * blockDim.x + threadIdx.x) >> 5;
  const int lane = threadIdx.x & 31;
  const int CT = DOUT / 32;                       // 8 column tiles of width 32
  const int total = (N_NODES / 16) * CT;          // 3125 * 8 = 25000 tiles
  if (wave >= total) return;                      // wave-uniform
  const int rt = wave / CT;
  const int ct = wave - rt * CT;

  const int mn = lane & 15;                       // A row / B col within tile
  const int kb = (lane >> 4) << 3;                // K base: 0 or 8

  const bf16* arow   = hbf + (size_t)(rt * 16 + mn) * DIN + kb;
  const bf16* browLo = wT  + (size_t)(ct * 32 + mn)      * DIN + kb;
  const bf16* browHi = wT  + (size_t)(ct * 32 + 16 + mn) * DIN + kb;

  v8f accLo = {}, accHi = {};
#pragma unroll
  for (int k0 = 0; k0 < DIN; k0 += 32) {
    FragAB a  = load_frag(arow + k0);
    FragAB bl = load_frag(browLo + k0);
    accLo = wmma_bf16(a, bl, accLo);
    FragAB bh = load_frag(browHi + k0);
    accHi = wmma_bf16(a, bh, accHi);
  }

  // C layout: lanes 0-15 -> M=r, N=lane; lanes 16-31 -> M=8+r, N=lane-16
  const float bvLo = bias[ct * 32 + mn];
  const float bvHi = bias[ct * 32 + 16 + mn];
  const int mbase = rt * 16 + ((lane >> 4) << 3);
#pragma unroll
  for (int r = 0; r < 8; ++r) {
    out[(size_t)(mbase + r) * DOUT + ct * 32 + mn]      = accLo[r] + bvLo;
    out[(size_t)(mbase + r) * DOUT + ct * 32 + 16 + mn] = accHi[r] + bvHi;
  }
}

// ---- edge term: prod_t sigmoid(concat(h[pairs_t], deg_t) @ W + b), scatter-add ----
// One wave: 16 edges x 32 output columns. A fragments (gathered h rows for
// t=0 and t=1) are loaded once per K-step and reused by both column tiles;
// B fragments shared across both t (same W per key). 4 wmma per K-step,
// 8 b128 loads per lane -> ~16 FLOP/B from cache.

__global__ void __launch_bounds__(256)
edge_term_wmma(const bf16* __restrict__ hbf,       // [N, DIN] bf16
               const int* __restrict__ pairs,      // [T=2, E] int32
               const float* __restrict__ degrees,  // [T=2, E, OH] fp32
               const int* __restrict__ scatter,    // [E] int32
               const bf16* __restrict__ wT,        // [DOUT, KPAD] bf16 (zero-padded W^T)
               const float* __restrict__ bias,     // [DOUT]
               const float* __restrict__ eps,      // [1]
               float* __restrict__ out) {          // [N, DOUT] fp32 (atomic +=)
  const int wave = (blockIdx.x * blockDim.x + threadIdx.x) >> 5;
  const int lane = threadIdx.x & 31;
  const int CT = DOUT / 32;                        // 8 column tiles of width 32
  const int total = (E_EDGES / 16) * CT;           // 15625 * 8 = 125000
  if (wave >= total) return;                       // wave-uniform
  const int rt = wave / CT;
  const int ct = wave - rt * CT;

  const int mn = lane & 15;
  const int kb = (lane >> 4) << 3;
  const int e  = rt * 16 + mn;                     // this lane's edge (A row)

  const int i0 = pairs[e];                         // t=0 gather node
  const int i1 = pairs[E_EDGES + e];               // t=1 gather node
  const bf16* a0row  = hbf + (size_t)i0 * DIN + kb;
  const bf16* a1row  = hbf + (size_t)i1 * DIN + kb;
  const bf16* browLo = wT + (size_t)(ct * 32 + mn)      * KPAD + kb;
  const bf16* browHi = wT + (size_t)(ct * 32 + 16 + mn) * KPAD + kb;

  v8f acc0Lo = {}, acc0Hi = {}, acc1Lo = {}, acc1Hi = {};
#pragma unroll
  for (int k0 = 0; k0 < DIN; k0 += 32) {
    FragAB bl = load_frag(browLo + k0);
    FragAB bh = load_frag(browHi + k0);
    FragAB a0 = load_frag(a0row + k0);
    acc0Lo = wmma_bf16(a0, bl, acc0Lo);
    acc0Hi = wmma_bf16(a0, bh, acc0Hi);
    FragAB a1 = load_frag(a1row + k0);
    acc1Lo = wmma_bf16(a1, bl, acc1Lo);
    acc1Hi = wmma_bf16(a1, bh, acc1Hi);
  }

  // Final K block (K=256..287): degrees in K 256..271, zeros 272..287.
  // Fragment elements [0..7] map to rel-K kb..kb+7 (valid deg), [8..15] to
  // rel-K >=16 -> zero; matching zero-padded W rows.
  {
    FragAB bl = load_frag(browLo + DIN);
    FragAB bh = load_frag(browHi + DIN);
    FragAB a0, a1;
    const float* d0 = degrees + (size_t)e * OH + kb;
    const float* d1 = degrees + ((size_t)E_EDGES + e) * OH + kb;
#pragma unroll
    for (int i = 0; i < 8; ++i) {
      a0.v[i] = (bf16)d0[i];
      a1.v[i] = (bf16)d1[i];
      a0.v[i + 8] = (bf16)0.0f;
      a1.v[i + 8] = (bf16)0.0f;
    }
    acc0Lo = wmma_bf16(a0, bl, acc0Lo);
    acc0Hi = wmma_bf16(a0, bh, acc0Hi);
    acc1Lo = wmma_bf16(a1, bl, acc1Lo);
    acc1Hi = wmma_bf16(a1, bh, acc1Hi);
  }

  const float bvLo  = bias[ct * 32 + mn];
  const float bvHi  = bias[ct * 32 + 16 + mn];
  const float scale = 1.0f + eps[0];

  v8f prodLo, prodHi;
#pragma unroll
  for (int r = 0; r < 8; ++r) {
    prodLo[r] = sigmoidf_fast(acc0Lo[r] + bvLo) * sigmoidf_fast(acc1Lo[r] + bvLo) * scale;
    prodHi[r] = sigmoidf_fast(acc0Hi[r] + bvHi) * sigmoidf_fast(acc1Hi[r] + bvHi) * scale;
  }

  // segment_sum: lane owns columns ct*32+mn and ct*32+16+mn for rows mbase..mbase+7
  const int mbase = rt * 16 + ((lane >> 4) << 3);
#pragma unroll
  for (int r = 0; r < 8; ++r) {
    const int dst = scatter[mbase + r];
    float* drow = out + (size_t)dst * DOUT + ct * 32 + mn;
    atomicAdd(drow, prodLo[r]);
    atomicAdd(drow + 16, prodHi[r]);
  }
}

// ---------------- launch ----------------

extern "C" void kernel_launch(void* const* d_in, const int* in_sizes, int n_in,
                              void* d_out, int out_size, void* d_ws, size_t ws_size,
                              hipStream_t stream) {
  const float* h        = (const float*)d_in[0];
  const int*   pairs_k0 = (const int*)  d_in[1];
  const int*   pairs_k1 = (const int*)  d_in[2];
  const float* deg_k0   = (const float*)d_in[3];
  const float* deg_k1   = (const float*)d_in[4];
  const int*   sc_k0    = (const int*)  d_in[5];
  const int*   sc_k1    = (const int*)  d_in[6];
  const float* W_lin    = (const float*)d_in[7];
  const float* b_lin    = (const float*)d_in[8];
  const float* W_k0     = (const float*)d_in[9];
  const float* b_k0     = (const float*)d_in[10];
  const float* W_k1     = (const float*)d_in[11];
  const float* b_k1     = (const float*)d_in[12];
  const float* eps_k0   = (const float*)d_in[13];
  const float* eps_k1   = (const float*)d_in[14];
  float* out = (float*)d_out;

  // Workspace layout (all offsets 16B aligned)
  char* ws = (char*)d_ws;
  bf16* hbf   = (bf16*)(ws);                                       // N*DIN   bf16
  bf16* wlinT = (bf16*)(ws + (size_t)N_NODES * DIN * 2);           // DOUT*DIN
  bf16* wk0T  = (bf16*)((char*)wlinT + (size_t)DOUT * DIN * 2);    // DOUT*KPAD
  bf16* wk1T  = (bf16*)((char*)wk0T + (size_t)DOUT * KPAD * 2);    // DOUT*KPAD

  // 1) pack h and transposed/padded weights to bf16
  {
    int n4 = N_NODES * DIN / 4;
    pack_h_bf16<<<(n4 + 255) / 256, 256, 0, stream>>>(h, hbf, n4);
    int tl = DOUT * DIN;
    pack_wT_bf16<<<(tl + 255) / 256, 256, 0, stream>>>(W_lin, wlinT, DIN, DIN, DOUT);
    int tk = DOUT * KPAD;
    pack_wT_bf16<<<(tk + 255) / 256, 256, 0, stream>>>(W_k0, wk0T, DIN + OH, KPAD, DOUT);
    pack_wT_bf16<<<(tk + 255) / 256, 256, 0, stream>>>(W_k1, wk1T, DIN + OH, KPAD, DOUT);
  }

  // 2) h3 = h @ W_lin + b_lin  (writes every element of d_out)
  // 25000 wave-tiles (16x32), 8 waves/block -> 3125 blocks exactly
  node_linear_wmma<<<3125, 256, 0, stream>>>(hbf, wlinT, b_lin, out);

  // 3) += (1+eps_k) * segment_sum(prod_t sigmoid(...)) for both keys
  // 125000 wave-tiles (16x32), 8 waves/block -> 15625 blocks exactly
  edge_term_wmma<<<15625, 256, 0, stream>>>(hbf, pairs_k0, deg_k0, sc_k0,
                                            wk0T, b_k0, eps_k0, out);
  edge_term_wmma<<<15625, 256, 0, stream>>>(hbf, pairs_k1, deg_k1, sc_k1,
                                            wk1T, b_k1, eps_k1, out);
}
